// BSNN_80547816669388
// MI455X (gfx1250) — 8-bit matrix operand order repaired and checked by reading, compiled, not run on hardware
//
#include <hip/hip_runtime.h>
#include <stdint.h>

typedef int v8i __attribute__((ext_vector_type(8)));

#define IMG       4
#define NBLK      (4096 / IMG)
#define NTHREADS  256

#define T1   2.3599835635698114f
#define T2   7.985043705972782f
#define T3   3.849629060468402f
#define BETA 0.44154740154430405f

#define OFF_BW2   0
#define OFF_BWFC  6144
#define OFF_SW1   14144
#define OFF_BN1S  14720
#define OFF_BN1B  14784
#define OFF_BN2S  14848
#define OFF_BN2B  14976
#define OFF_MEM3  15104
#define OFF_CUR1  15264
#define OFF_MEM1  58528
#define OFF_SPK1  101792
#define OFF_MEM2  112608
#define OFF_SPK2  125408
#define OFF_CONV  128608
#define SMEM_BYTES 192096

__device__ __forceinline__ float fsign(float x) {
    return (x > 0.f) ? 1.f : ((x < 0.f) ? -1.f : 0.f);
}
__device__ __forceinline__ signed char s8sign(float x) {
    return (x > 0.f) ? (signed char)1 : ((x < 0.f) ? (signed char)-1 : (signed char)0);
}

__global__ __launch_bounds__(NTHREADS)
void bsnn_fused_kernel(const float* __restrict__ inpt, const float* __restrict__ w1,
                       const float* __restrict__ w2,  const float* __restrict__ w_fc,
                       const float* __restrict__ b_fc,
                       const float* __restrict__ bn1g, const float* __restrict__ bn1b,
                       const float* __restrict__ bn1m, const float* __restrict__ bn1v,
                       const float* __restrict__ bn2g, const float* __restrict__ bn2b,
                       const float* __restrict__ bn2m, const float* __restrict__ bn2v,
                       float* __restrict__ out)
{
    extern __shared__ char smem[];
    signed char*   sBW2  = (signed char*)  (smem + OFF_BW2);
    signed char*   sBWFC = (signed char*)  (smem + OFF_BWFC);
    float*         sSW1  = (float*)        (smem + OFF_SW1);
    float*         sBN1S = (float*)        (smem + OFF_BN1S);
    float*         sBN1B = (float*)        (smem + OFF_BN1B);
    float*         sBN2S = (float*)        (smem + OFF_BN2S);
    float*         sBN2B = (float*)        (smem + OFF_BN2B);
    float*         sMEM3 = (float*)        (smem + OFF_MEM3);
    float*         sCUR1 = (float*)        (smem + OFF_CUR1);
    float*         sMEM1 = (float*)        (smem + OFF_MEM1);
    unsigned char* sSPK1 = (unsigned char*)(smem + OFF_SPK1);
    float*         sMEM2 = (float*)        (smem + OFF_MEM2);
    unsigned char* sSPK2 = (unsigned char*)(smem + OFF_SPK2);
    float*         sCONV = (float*)        (smem + OFF_CONV);

    const int tid  = threadIdx.x;
    const int lane = tid & 31;
    const int wid  = tid >> 5;
    const int half = lane >> 4;
    const int nl   = lane & 15;

    for (int i = tid; i < 32 * 192; i += NTHREADS) {
        int n = i / 192, k = i - n * 192;
        signed char v = 0;
        if (k < 144) {
            int c = k & 15, tap = k >> 4;
            int ky = tap / 3, kx = tap - 3 * ky;
            v = s8sign(w2[((n * 16 + c) * 3 + ky) * 3 + kx]);
        }
        sBW2[i] = v;
    }
    for (int i = tid; i < 8000; i += NTHREADS) sBWFC[i] = s8sign(w_fc[i]);
    for (int i = tid; i < 144;  i += NTHREADS) sSW1[i]  = fsign(w1[i]);
    for (int i = tid; i < 16;   i += NTHREADS) {
        float sc = bn1g[i] * rsqrtf(bn1v[i] + 1e-5f);
        sBN1S[i] = sc; sBN1B[i] = bn1b[i] - bn1m[i] * sc;
    }
    for (int i = tid; i < 32;   i += NTHREADS) {
        float sc = bn2g[i] * rsqrtf(bn2v[i] + 1e-5f);
        sBN2S[i] = sc; sBN2B[i] = bn2b[i] - bn2m[i] * sc;
    }
    for (int i = tid; i < IMG * 2704; i += NTHREADS) sMEM1[i] = 0.f;
    for (int i = tid; i < IMG * 800;  i += NTHREADS) sMEM2[i] = 0.f;
    for (int i = tid; i < IMG * 10;   i += NTHREADS) sMEM3[i] = 0.f;

    float* sINP = sCONV;
    for (int i = tid; i < IMG * 784; i += NTHREADS)
        sINP[i] = inpt[(size_t)blockIdx.x * IMG * 784 + i];
    __syncthreads();

    for (int i = tid; i < IMG * 2704; i += NTHREADS) {
        int imgL = i / 2704, rem = i - imgL * 2704;
        int pix = rem >> 4, c = rem & 15;
        int py = pix / 13, px = pix - 13 * py;
        float best = -3.402823466e38f;
#pragma unroll
        for (int dy = 0; dy < 2; ++dy)
#pragma unroll
        for (int dx = 0; dx < 2; ++dx) {
            int y0 = 2 * py + dy, x0 = 2 * px + dx;
            float s = 0.f;
#pragma unroll
            for (int ky = 0; ky < 3; ++ky)
#pragma unroll
            for (int kx = 0; kx < 3; ++kx)
                s += sINP[imgL * 784 + (y0 + ky) * 28 + (x0 + kx)] * sSW1[c * 9 + ky * 3 + kx];
            best = fmaxf(best, s * sBN1S[c] + sBN1B[c]);
        }
        sCUR1[i] = best;
    }
    __syncthreads();

    v8i Bf[6];
#pragma unroll
    for (int nt = 0; nt < 2; ++nt)
#pragma unroll
    for (int ch = 0; ch < 3; ++ch) {
        v8i b;
#pragma unroll
        for (int v = 0; v < 8; ++v) {
            int kb = ch * 64 + ((v >> 2) << 5) + (half << 4) + ((v & 3) << 2);
            b[v] = *(const int*)(sBW2 + (nt * 16 + nl) * 192 + kb);
        }
        Bf[nt * 3 + ch] = b;
    }

    for (int t = 0; t < 10; ++t) {
        for (int i = tid; i < IMG * 2704; i += NTHREADS) {
            float m  = sMEM1[i];
            float mn = BETA * m + sCUR1[i] - ((m > T1) ? T1 : 0.f);
            sMEM1[i] = mn;
            sSPK1[i] = (unsigned char)(((mn - T1) > 0.f) ? 1 : 0);
        }
        __syncthreads();

        for (int tm = wid; tm < 31; tm += 8) {
            int r = tm * 16 + nl;
            bool valid = r < IMG * 121;
            int rc  = valid ? r : 0;
            int img = rc / 121, pos = rc - img * 121;
            int oy  = pos / 11,  ox = pos - oy * 11;
            int abase = (img * 169 + oy * 13 + ox) * 16;

            v8i acc0 = {0, 0, 0, 0, 0, 0, 0, 0};
            v8i acc1 = {0, 0, 0, 0, 0, 0, 0, 0};
#pragma unroll
            for (int ch = 0; ch < 3; ++ch) {
                v8i a;
#pragma unroll
                for (int v = 0; v < 8; ++v) {
                    const int tap = ch * 4 + ((v >> 2) << 1) + half;
                    int val = 0;
                    if (tap < 9) {
                        const int ky = tap / 3, kx = tap - 3 * (tap / 3);
                        int c0 = (v & 3) << 2;
                        if (valid)
                            val = *(const int*)(sSPK1 + abase + (ky * 13 + kx) * 16 + c0);
                    }
                    a[v] = val;
                }
                acc0 = __builtin_amdgcn_wmma_i32_16x16x64_iu8(false, a, true, Bf[ch],     acc0, false, false);
                acc1 = __builtin_amdgcn_wmma_i32_16x16x64_iu8(false, a, true, Bf[3 + ch], acc1, false, false);
            }
#pragma unroll
            for (int d = 0; d < 8; ++d) {
                int rr = tm * 16 + (half << 3) + d;
                sCONV[rr * 32 + nl]      = (float)acc0[d] * sBN2S[nl]      + sBN2B[nl];
                sCONV[rr * 32 + 16 + nl] = (float)acc1[d] * sBN2S[16 + nl] + sBN2B[16 + nl];
            }
        }
        __syncthreads();

        for (int i = tid; i < IMG * 800; i += NTHREADS) {
            int imgL = i / 800, rem = i - imgL * 800;
            int chn = rem / 25, p = rem - chn * 25;
            int py = p / 5, px = p - 5 * py;
            int r0 = imgL * 121 + (2 * py) * 11 + 2 * px;
            float c2 = fmaxf(fmaxf(sCONV[r0 * 32 + chn],        sCONV[(r0 + 1) * 32 + chn]),
                             fmaxf(sCONV[(r0 + 11) * 32 + chn], sCONV[(r0 + 12) * 32 + chn]));
            float m  = sMEM2[i];
            float mn = BETA * m + c2 - ((m > T2) ? T2 : 0.f);
            sMEM2[i] = mn;
            sSPK2[i] = (unsigned char)(((mn - T2) > 0.f) ? 1 : 0);
        }
        __syncthreads();

        for (int job = wid; job < IMG * 10; job += 8) {
            int imgL = job / 10, j = job - imgL * 10;
            int acc = 0;
            for (int k = lane; k < 800; k += 32)
                acc += (int)sSPK2[imgL * 800 + k] * (int)sBWFC[j * 800 + k];
#pragma unroll
            for (int off = 16; off > 0; off >>= 1)
                acc += __shfl_xor(acc, off, 32);
            if (lane == 0) {
                float cur3 = (float)acc + b_fc[j];
                float m  = sMEM3[job];
                float mn = BETA * m + cur3 - ((m > T3) ? T3 : 0.f);
                sMEM3[job] = mn;
                float spk = ((mn - T3) > 0.f) ? 1.f : 0.f;
                size_t b = (size_t)blockIdx.x * IMG + imgL;
                out[((size_t)t * 4096 + b) * 10 + j]          = spk;
                out[409600 + ((size_t)t * 4096 + b) * 10 + j] = mn;
            }
        }
        __syncthreads();
    }
}

extern "C" void kernel_launch(void* const* d_in, const int* in_sizes, int n_in,
                              void* d_out, int out_size, void* d_ws, size_t ws_size,
                              hipStream_t stream) {
    (void)in_sizes; (void)n_in; (void)out_size; (void)d_ws; (void)ws_size;
    const float* inpt = (const float*)d_in[0];
    const float* w1   = (const float*)d_in[1];
    const float* w2   = (const float*)d_in[2];
    const float* w_fc = (const float*)d_in[3];
    const float* b_fc = (const float*)d_in[4];
    const float* bn1g = (const float*)d_in[5];
    const float* bn1b = (const float*)d_in[6];
    const float* bn1m = (const float*)d_in[7];
    const float* bn1v = (const float*)d_in[8];
    const float* bn2g = (const float*)d_in[9];
    const float* bn2b = (const float*)d_in[10];
    const float* bn2m = (const float*)d_in[11];
    const float* bn2v = (const float*)d_in[12];

    static int attr_done = 0;
    if (!attr_done) {
        hipFuncSetAttribute((const void*)bsnn_fused_kernel,
                            hipFuncAttributeMaxDynamicSharedMemorySize, SMEM_BYTES);
        attr_done = 1;
    }

    bsnn_fused_kernel<<<dim3(NBLK), dim3(NTHREADS), SMEM_BYTES, stream>>>(
        inpt, w1, w2, w_fc, b_fc,
        bn1g, bn1b, bn1m, bn1v,
        bn2g, bn2b, bn2m, bn2v,
        (float*)d_out);
}
